// PerformerMelRefinementNetwork_58085137711605
// MI455X (gfx1250) — matrix operand order repaired and checked by reading, not compiled, not run on hardware
//
#include <hip/hip_runtime.h>
#include <hip/hip_bf16.h>
#include <math.h>

// ---------------------------------------------------------------------------
// PerformerMelRefinementNetwork forward for MI455X (gfx1250, wave32, WMMA).
// Core: generic f32->f16 WMMA GEMM (v_wmma_f32_16x16x32_f16) with fused
// epilogues (bias / relu / gelu / row-scale / residual / batching).
// BK=64 K-slab: 8 WMMAs per wave per barrier pair.
// ---------------------------------------------------------------------------

typedef _Float16 half_t;
typedef __attribute__((ext_vector_type(16))) _Float16 v16h;
typedef __attribute__((ext_vector_type(8)))  _Float16 v8h;
typedef __attribute__((ext_vector_type(4)))  _Float16 v4h;
typedef __attribute__((ext_vector_type(2)))  _Float16 v2h;
typedef __attribute__((ext_vector_type(8)))  float    v8f;
typedef __attribute__((ext_vector_type(4)))  float    v4f;

#define DIM_B  4
#define DIM_T  1920
#define DIM_NM 80
#define DIM_D  512
#define DIM_H  8
#define DIM_DH 64
#define DIM_FD 32
#define DIM_CD 153

enum { EP_BIAS = 1, EP_RELU = 2, EP_GELU = 4, EP_RES = 8, EP_ROWSCALE = 16 };

// ---------------------------------------------------------------------------
// WMMA helpers
// ---------------------------------------------------------------------------
__device__ __forceinline__ v8f wmma_f16(v16h a, v16h b, v8f c) {
  // 8 args: (neg_a, A, neg_b, B, c_mod, C, reuse_a, reuse_b)
  return __builtin_amdgcn_wmma_f32_16x16x32_f16(false, a, false, b, (short)0, c,
                                                false, false);
}

// A fragment (16x32 f16): lane group kg=lane/16 holds K = kg*8+0..7 and
// K = 16+kg*8+0..7 (interleaved per ISA table). LDS rows are 64 halves wide;
// rowbase already includes the k-substep column offset.
__device__ __forceinline__ v16h ldfragA(const half_t* rowbase, int kg) {
  v8h lo = *(const v8h*)(rowbase + kg * 8);
  v8h hi = *(const v8h*)(rowbase + kg * 8 + 16);
  return __builtin_shufflevector(lo, hi, 0, 1, 2, 3, 4, 5, 6, 7,
                                 8, 9, 10, 11, 12, 13, 14, 15);
}

// B fragment (32x16 f16): lane group kg holds K = kg*8+0..7 and K = 16+kg*8+0..7.
// B is stored transposed in LDS: [n][k], rows 64 halves wide.
__device__ __forceinline__ v16h ldfragB(const half_t* rowbase, int kg) {
  v8h lo = *(const v8h*)(rowbase + kg * 8);
  v8h hi = *(const v8h*)(rowbase + kg * 8 + 16);
  return __builtin_shufflevector(lo, hi, 0, 1, 2, 3, 4, 5, 6, 7,
                                 8, 9, 10, 11, 12, 13, 14, 15);
}

// ---------------------------------------------------------------------------
// Generic WMMA GEMM: C[M,N] = epilogue(A[M,K] * W[K,N]); f32 data, f16 MACs.
// Tile: 64(M) x 128(N) per workgroup; 8 waves, each 32x32 (2x2 WMMA tiles).
// K staged in 64-wide slabs (2 WMMA K-steps per barrier pair).
// blockIdx.z batches (strides in elements).
// Staging uses a scalar-uniform fast path (vectorized, unguarded) and a
// branchless clamp+select slow path for edge tiles — no EXEC-mask branching.
// ---------------------------------------------------------------------------
__global__ __launch_bounds__(256) void k_gemm(
    const float* __restrict__ Ag, long long strA, long long lda,
    const float* __restrict__ Wg, long long strB, long long ldb,
    float* __restrict__ Cg, long long strC, long long ldc,
    const float* __restrict__ bias,
    const float* __restrict__ resg, long long strR, long long ldr,
    const float* __restrict__ rsg, long long strS,
    int M, int N, int K, int flags, float alpha) {
  const int tid  = threadIdx.x;
  const int lane = tid & 31;
  const int wv   = tid >> 5;
  const long long zb = blockIdx.z;

  const float* Ap = Ag + zb * strA;
  const float* Wp = Wg + zb * strB;
  float*       Cp = Cg + zb * strC;
  const float* Rp = resg ? (resg + zb * strR) : nullptr;
  const float* Sp = rsg ? (rsg + zb * strS) : nullptr;

  const int m0 = blockIdx.y * 64;
  const int n0 = blockIdx.x * 128;
  const int wm = (wv >> 2) * 32;   // wave row offset in tile (0/32)
  const int wn = (wv & 3) * 32;    // wave col offset in tile (0..96)

  __shared__ half_t shA[64 * 64];    // [m][k], 8 KB
  __shared__ half_t shB[128 * 64];   // transposed: [n][k], 16 KB

  v8f acc[2][2];
#pragma unroll
  for (int mi = 0; mi < 2; ++mi)
#pragma unroll
    for (int ni = 0; ni < 2; ++ni)
#pragma unroll
      for (int e = 0; e < 8; ++e) acc[mi][ni][e] = 0.f;

  const int kg = lane >> 4;
  const int lr = lane & 15;

  const bool a_rows_full = (m0 + 64) <= M && ((lda & 3) == 0);
  const bool b_cols_full = (n0 + 128) <= N && ((ldb & 3) == 0);

  for (int k0 = 0; k0 < K; k0 += 64) {
    const bool kfull = (k0 + 64) <= K;   // scalar-uniform

    // ---- stage A slab (64x64), f32 -> f16 ----
    if (a_rows_full && kfull) {
#pragma unroll
      for (int i = 0; i < 4; ++i) {
        int idx = tid + i * 256;          // 0..1023
        int mm = idx >> 4;                // 0..63
        int kk = (idx & 15) * 4;          // 0,4,..,60
        const float* ap = Ap + (long long)(m0 + mm) * lda + (k0 + kk);
        v4f f = *(const v4f*)ap;
        if (k0 + 64 < K) __builtin_prefetch(ap + 64, 0, 1);
        v4h h;
        h[0] = (half_t)f[0]; h[1] = (half_t)f[1];
        h[2] = (half_t)f[2]; h[3] = (half_t)f[3];
        *(v4h*)&shA[mm * 64 + kk] = h;
      }
    } else {
#pragma unroll
      for (int i = 0; i < 16; ++i) {
        int idx = tid + i * 256;          // 0..4095
        int mm = idx >> 6, kk = idx & 63;
        int gm = m0 + mm, gk = k0 + kk;
        bool ok = (gm < M) && (gk < K);
        long long ofs = ok ? ((long long)gm * lda + gk) : 0;
        float v = Ap[ofs];
        shA[idx] = (half_t)(ok ? v : 0.f);
      }
    }

    // ---- stage B slab (64x128) transposed into [n][k] ----
    if (b_cols_full && kfull) {
#pragma unroll
      for (int i = 0; i < 4; ++i) {
        int idx = tid + i * 256;          // 0..1023
        int k2 = idx >> 5;                // k-pair id 0..31
        int nn = (idx & 31) * 4;          // 0..124
        const float* bp = Wp + (long long)(k0 + 2 * k2) * ldb + (n0 + nn);
        v4f f0 = *(const v4f*)bp;
        v4f f1 = *(const v4f*)(bp + ldb);
        if (k0 + 64 < K) __builtin_prefetch(bp + 64 * ldb, 0, 1);
#pragma unroll
        for (int j = 0; j < 4; ++j) {
          v2h p;
          p[0] = (half_t)f0[j];
          p[1] = (half_t)f1[j];
          *(v2h*)&shB[(nn + j) * 64 + 2 * k2] = p;
        }
      }
    } else {
#pragma unroll
      for (int i = 0; i < 32; ++i) {
        int idx = tid + i * 256;          // 0..8191
        int kk = idx >> 7, nn = idx & 127;
        int gk = k0 + kk, gn = n0 + nn;
        bool ok = (gk < K) && (gn < N);
        long long ofs = ok ? ((long long)gk * ldb + gn) : 0;
        float v = Wp[ofs];
        shB[nn * 64 + kk] = (half_t)(ok ? v : 0.f);
      }
    }
    __syncthreads();

    // ---- 2 K-substeps x 4 WMMAs within one barrier interval ----
#pragma unroll
    for (int ks = 0; ks < 2; ++ks) {
      const int kc = ks * 32;
      v16h af[2], bf[2];
      af[0] = ldfragA(&shA[(wm + lr) * 64 + kc], kg);
      af[1] = ldfragA(&shA[(wm + 16 + lr) * 64 + kc], kg);
      bf[0] = ldfragB(&shB[(wn + lr) * 64 + kc], kg);
      bf[1] = ldfragB(&shB[(wn + 16 + lr) * 64 + kc], kg);
#pragma unroll
      for (int mi = 0; mi < 2; ++mi)
#pragma unroll
        for (int ni = 0; ni < 2; ++ni)
          acc[mi][ni] = wmma_f16(af[mi], bf[ni], acc[mi][ni]);
    }

    __syncthreads();
  }

  // ---- epilogue & store: C VGPR r -> row r + 8*(lane/16), col lane%16 ----
#pragma unroll
  for (int mi = 0; mi < 2; ++mi) {
#pragma unroll
    for (int ni = 0; ni < 2; ++ni) {
      int gn = n0 + wn + ni * 16 + lr;
      if (gn >= N) continue;
      float bv = (flags & EP_BIAS) ? bias[gn] : 0.f;
#pragma unroll
      for (int r = 0; r < 8; ++r) {
        int gm = m0 + wm + mi * 16 + kg * 8 + r;
        if (gm >= M) continue;
        float v = acc[mi][ni][r] + bv;
        if (flags & EP_RELU) v = fmaxf(v, 0.f);
        if (flags & EP_GELU) v = 0.5f * v * (1.f + erff(v * 0.70710678f));
        v *= alpha;
        if (flags & EP_ROWSCALE) v *= Sp[gm];
        if (flags & EP_RES) v += Rp[(long long)gm * ldr + gn];
        Cp[(long long)gm * ldc + gn] = v;
      }
    }
  }
}

// ---------------------------------------------------------------------------
// Small support kernels
// ---------------------------------------------------------------------------

// g[b, c] = mean over T of concat([f0, phoneme, singer, language], -1)
__global__ void k_condmean(const float* __restrict__ f0,
                           const float* __restrict__ ph,
                           const float* __restrict__ sg,
                           const float* __restrict__ lg,
                           float* __restrict__ g) {
  int i = blockIdx.x * blockDim.x + threadIdx.x;
  if (i >= DIM_B * DIM_CD) return;
  int b = i / DIM_CD, c = i % DIM_CD;
  float s = 0.f;
  if (c == 0) {
    for (int t = 0; t < DIM_T; ++t) s += f0[(size_t)b * DIM_T + t];
    s *= (1.f / DIM_T);
  } else if (c < 129) {
    int j = c - 1;
    for (int t = 0; t < DIM_T; ++t)
      s += ph[((size_t)b * DIM_T + t) * 128 + j];
    s *= (1.f / DIM_T);
  } else if (c < 145) {
    s = sg[b * 16 + (c - 129)];
  } else {
    s = lg[b * 8 + (c - 145)];
  }
  g[i] = s;
}

// LayerNorm over last dim (one workgroup per row)
__global__ __launch_bounds__(256) void k_layernorm(
    const float* __restrict__ x, const float* __restrict__ g,
    const float* __restrict__ be, float* __restrict__ y, int Dd) {
  __shared__ float red[256];
  int row = blockIdx.x, tid = threadIdx.x;
  const float* xr = x + (size_t)row * Dd;
  float s = 0.f;
  for (int c = tid; c < Dd; c += 256) s += xr[c];
  red[tid] = s; __syncthreads();
  for (int o = 128; o > 0; o >>= 1) {
    if (tid < o) red[tid] += red[tid + o];
    __syncthreads();
  }
  float mu = red[0] / Dd;
  __syncthreads();
  float v = 0.f;
  for (int c = tid; c < Dd; c += 256) { float d = xr[c] - mu; v += d * d; }
  red[tid] = v; __syncthreads();
  for (int o = 128; o > 0; o >>= 1) {
    if (tid < o) red[tid] += red[tid + o];
    __syncthreads();
  }
  float rs = rsqrtf(red[0] / Dd + 1e-5f);
  float* yr = y + (size_t)row * Dd;
  for (int c = tid; c < Dd; c += 256) yr[c] = (xr[c] - mu) * rs * g[c] + be[c];
}

// x += positional encoding (sin/cos interleave)
__global__ void k_posenc(float* __restrict__ x, int Mtot, int Tn) {
  int i = blockIdx.x * blockDim.x + threadIdx.x;
  if (i >= Mtot * DIM_D) return;
  int row = i / DIM_D, c = i % DIM_D;
  int t = row % Tn;
  float e = (float)((c >> 1) << 1) * (-9.210340372f / DIM_D); // -ln(1e4)/D
  float ang = (float)t * expf(e);
  x[i] += (c & 1) ? cosf(ang) : sinf(ang);
}

// ctx[b*Ntok + t, :] = cbc[b, :]
__global__ void k_broadcast(const float* __restrict__ cbc,
                            float* __restrict__ ctx, int Ntok) {
  int i = blockIdx.x * blockDim.x + threadIdx.x;
  if (i >= DIM_B * Ntok * DIM_D) return;
  int row = i / DIM_D, c = i % DIM_D;
  int b = row / Ntok;
  ctx[i] = cbc[b * DIM_D + c];
}

// kv[(h*B+b), f, d] = sum_m kp[h, b*N+m, f] * v[b*N+m, h*64+d]; ksum likewise
__global__ __launch_bounds__(256) void k_kvsum(
    const float* __restrict__ kp, const float* __restrict__ v,
    float* __restrict__ kv, float* __restrict__ ksum, int Ntok, int Mtot) {
  int h = blockIdx.x / DIM_B, b = blockIdx.x % DIM_B;
  int tid = threadIdx.x;
  float acc[8];
#pragma unroll
  for (int j = 0; j < 8; ++j) acc[j] = 0.f;
  float ks = 0.f;
  for (int m = 0; m < Ntok; ++m) {
    size_t kprow = ((size_t)h * Mtot + (size_t)b * Ntok + m) * DIM_FD;
    size_t vrow  = ((size_t)b * Ntok + m) * DIM_D + h * DIM_DH;
#pragma unroll
    for (int j = 0; j < 8; ++j) {
      int e = tid + j * 256;
      int f = e >> 6, d = e & 63;
      acc[j] += kp[kprow + f] * v[vrow + d];
    }
    if (tid < DIM_FD) ks += kp[kprow + tid];
  }
  size_t base = (size_t)blockIdx.x * (DIM_FD * DIM_DH);
#pragma unroll
  for (int j = 0; j < 8; ++j) kv[base + tid + j * 256] = acc[j];
  if (tid < DIM_FD) ksum[(size_t)blockIdx.x * DIM_FD + tid] = ks;
}

// z[h, row] = 1 / (qp[h,row,:] . ksum[h*B+b,:] + 1e-8)
__global__ void k_z(const float* __restrict__ qp,
                    const float* __restrict__ ksum,
                    float* __restrict__ z, int Mtot, int Ntok) {
  int i = blockIdx.x * blockDim.x + threadIdx.x;
  if (i >= DIM_H * Mtot) return;
  int h = i / Mtot, row = i % Mtot;
  int b = row / Ntok;
  const float* q  = qp + ((size_t)h * Mtot + row) * DIM_FD;
  const float* ks = ksum + (size_t)(h * DIM_B + b) * DIM_FD;
  float s = 0.f;
#pragma unroll
  for (int f = 0; f < DIM_FD; ++f) s += q[f] * ks[f];
  z[(size_t)h * Mtot + row] = 1.f / (s + 1e-8f);
}

// MaxPool1d(k=2,s=2) over time
__global__ void k_maxpool(const float* __restrict__ x, float* __restrict__ y,
                          int T2, int W) {
  long long i = (long long)blockIdx.x * blockDim.x + threadIdx.x;
  long long tot = (long long)DIM_B * T2 * W;
  if (i >= tot) return;
  long long row = i / W; int c = (int)(i % W);
  int b = (int)(row / T2), l = (int)(row % T2);
  size_t r0 = ((size_t)b * (2 * T2) + 2 * l) * W + c;
  y[i] = fmaxf(x[r0], x[r0 + W]);
}

// nearest x2 upsample over time (generic width)
__global__ void k_upsample(const float* __restrict__ x, float* __restrict__ y,
                           int Tout, int W) {
  long long i = (long long)blockIdx.x * blockDim.x + threadIdx.x;
  long long tot = (long long)DIM_B * Tout * W;
  if (i >= tot) return;
  long long row = i / W; int c = (int)(i % W);
  int b = (int)(row / Tout), t = (int)(row % Tout);
  y[i] = x[((size_t)b * (Tout / 2) + (t >> 1)) * W + c];
}

// dst[r, colOff + c] = src[r, c]
__global__ void k_copycols(const float* __restrict__ src, long long ldS,
                           float* __restrict__ dst, long long ldD,
                           int colOff, int Mrows, int Cc) {
  long long i = (long long)blockIdx.x * blockDim.x + threadIdx.x;
  if (i >= (long long)Mrows * Cc) return;
  long long r = i / Cc; int c = (int)(i % Cc);
  dst[r * ldD + colOff + c] = src[r * ldS + c];
}

// ---------------------------------------------------------------------------
// Host-side parameter plumbing and orchestration
// ---------------------------------------------------------------------------
struct Lin  { const float* w; const float* b; };
struct LN   { const float* g; const float* b; };
struct AttnP{ Lin q, k, v, o; const float* proj; };
struct Block{ LN n1; AttnP sa; LN n2; AttnP ca; LN n3; Lin ff1, ff2, cond; };

struct Temps {
  float *y, *q, *k, *v, *ctx, *attn, *ff;
  float *qp, *kp, *z, *ksum, *kv, *cbc;
  const float* cb;
};

static void gemm_full(hipStream_t st, const float* A, long long lda, long long sA,
                      const float* W, long long ldb, long long sB,
                      float* C, long long ldc, long long sC,
                      int M, int N, int K, int batch,
                      const float* bias, const float* res, long long ldr, long long sR,
                      const float* rowscale, long long sS, int flags, float alpha) {
  if (bias) flags |= EP_BIAS;
  if (res) flags |= EP_RES;
  if (rowscale) flags |= EP_ROWSCALE;
  dim3 g((N + 127) / 128, (M + 63) / 64, batch);
  k_gemm<<<g, dim3(256), 0, st>>>(A, sA, lda, W, sB, ldb, C, sC, ldc,
                                  bias, res, sR, ldr, rowscale, sS,
                                  M, N, K, flags, alpha);
}

static void gemm(hipStream_t st, const float* A, long long lda,
                 const float* W, long long ldb, float* C, long long ldc,
                 int M, int N, int K, const float* bias, int flags = 0,
                 float alpha = 1.f, const float* res = nullptr,
                 long long ldr = 0) {
  gemm_full(st, A, lda, 0, W, ldb, 0, C, ldc, 0, M, N, K, 1,
            bias, res, ldr, 0, nullptr, 0, flags, alpha);
}

static void copycols(hipStream_t st, const float* src, long long ldS,
                     float* dst, long long ldD, int colOff, int M, int Cc) {
  long long tot = (long long)M * Cc;
  k_copycols<<<dim3((unsigned)((tot + 255) / 256)), dim3(256), 0, st>>>(
      src, ldS, dst, ldD, colOff, M, Cc);
}

// Performer attention: xres += Wo * attn(qsrc, ctx) (residual fused)
static void attn(hipStream_t st, float* xres, const float* qsrc,
                 const float* ctx, int Mtot, int Ntok, const AttnP& p,
                 Temps& t) {
  gemm(st, qsrc, DIM_D, p.q.w, DIM_D, t.q, DIM_D, Mtot, DIM_D, DIM_D, p.q.b);
  gemm(st, ctx,  DIM_D, p.k.w, DIM_D, t.k, DIM_D, Mtot, DIM_D, DIM_D, p.k.b);
  gemm(st, ctx,  DIM_D, p.v.w, DIM_D, t.v, DIM_D, Mtot, DIM_D, DIM_D, p.v.b);

  const float isf = 0.17677669529f;  // 1/sqrt(FD)
  // qp/kp: relu(head_slice @ proj) * isf, batched over 8 heads (z-dim)
  gemm_full(st, t.q, DIM_D, 64, p.proj, DIM_FD, 0,
            t.qp, DIM_FD, (long long)Mtot * DIM_FD,
            Mtot, DIM_FD, DIM_DH, DIM_H, nullptr, nullptr, 0, 0,
            nullptr, 0, EP_RELU, isf);
  gemm_full(st, t.k, DIM_D, 64, p.proj, DIM_FD, 0,
            t.kp, DIM_FD, (long long)Mtot * DIM_FD,
            Mtot, DIM_FD, DIM_DH, DIM_H, nullptr, nullptr, 0, 0,
            nullptr, 0, EP_RELU, isf);

  k_kvsum<<<dim3(DIM_H * DIM_B), dim3(256), 0, st>>>(t.kp, t.v, t.kv, t.ksum,
                                                     Ntok, Mtot);
  int zt = DIM_H * Mtot;
  k_z<<<dim3((zt + 255) / 256), dim3(256), 0, st>>>(t.qp, t.ksum, t.z, Mtot,
                                                    Ntok);

  // out_head = (qp @ kv) * z, per head (batched over batch index b)
  for (int h = 0; h < DIM_H; ++h) {
    gemm_full(st,
              t.qp + (size_t)h * Mtot * DIM_FD, DIM_FD, (long long)Ntok * DIM_FD,
              t.kv + (size_t)h * DIM_B * DIM_FD * DIM_DH, DIM_DH,
              (long long)DIM_FD * DIM_DH,
              t.attn + h * DIM_DH, DIM_D, (long long)Ntok * DIM_D,
              Ntok, DIM_DH, DIM_FD, DIM_B, nullptr, nullptr, 0, 0,
              t.z + (size_t)h * Mtot, Ntok, 0, 1.f);
  }
  // output projection + residual into xres
  gemm(st, t.attn, DIM_D, p.o.w, DIM_D, xres, DIM_D, Mtot, DIM_D, DIM_D,
       p.o.b, 0, 1.f, xres, DIM_D);
}

static void run_block(hipStream_t st, float* x, int Mtot, int Ntok,
                      const Block& p, Temps& t) {
  // conditioning projection (tiny GEMM, 4 rows)
  gemm(st, t.cb, DIM_D, p.cond.w, DIM_D, t.cbc, DIM_D, DIM_B, DIM_D, DIM_D,
       p.cond.b);
  // self-attention
  k_layernorm<<<dim3(Mtot), dim3(256), 0, st>>>(x, p.n1.g, p.n1.b, t.y, DIM_D);
  attn(st, x, t.y, t.y, Mtot, Ntok, p.sa, t);
  // cross-attention with broadcast conditioning context
  k_layernorm<<<dim3(Mtot), dim3(256), 0, st>>>(x, p.n2.g, p.n2.b, t.y, DIM_D);
  {
    long long tot = (long long)Mtot * DIM_D;
    k_broadcast<<<dim3((unsigned)((tot + 255) / 256)), dim3(256), 0, st>>>(
        t.cbc, t.ctx, Ntok);
  }
  attn(st, x, t.y, t.ctx, Mtot, Ntok, p.ca, t);
  // feed-forward (GELU) with fused residual
  k_layernorm<<<dim3(Mtot), dim3(256), 0, st>>>(x, p.n3.g, p.n3.b, t.y, DIM_D);
  gemm(st, t.y, DIM_D, p.ff1.w, 4 * DIM_D, t.ff, 4 * DIM_D, Mtot, 4 * DIM_D,
       DIM_D, p.ff1.b, EP_GELU);
  gemm(st, t.ff, 4 * DIM_D, p.ff2.w, DIM_D, x, DIM_D, Mtot, DIM_D, 4 * DIM_D,
       p.ff2.b, 0, 1.f, x, DIM_D);
}

// ---------------------------------------------------------------------------
extern "C" void kernel_launch(void* const* d_in, const int* in_sizes, int n_in,
                              void* d_out, int out_size, void* d_ws,
                              size_t ws_size, hipStream_t stream) {
  (void)in_sizes; (void)n_in; (void)out_size; (void)ws_size;

  const float* mel = (const float*)d_in[0];
  const float* f0  = (const float*)d_in[1];
  const float* ph  = (const float*)d_in[2];
  const float* sg  = (const float*)d_in[3];
  const float* lg  = (const float*)d_in[4];

  // params flattened depth-first in dict insertion order
  int idx = 5;
  auto P = [&]() { return (const float*)d_in[idx++]; };
  Lin input_proj{P(), P()}, cond1{P(), P()}, cond2{P(), P()};
  Lin down1{P(), P()}, down2{P(), P()}, up1{P(), P()}, up2{P(), P()};
  Lin outp{P(), P()};
  Block blocks[12];
  for (int i = 0; i < 12; ++i) {
    Block& b = blocks[i];
    b.n1 = {P(), P()};
    b.sa.q = {P(), P()}; b.sa.k = {P(), P()}; b.sa.v = {P(), P()};
    b.sa.o = {P(), P()}; b.sa.proj = P();
    b.n2 = {P(), P()};
    b.ca.q = {P(), P()}; b.ca.k = {P(), P()}; b.ca.v = {P(), P()};
    b.ca.o = {P(), P()}; b.ca.proj = P();
    b.n3 = {P(), P()};
    b.ff1 = {P(), P()}; b.ff2 = {P(), P()}; b.cond = {P(), P()};
  }

  const int M1 = DIM_B * DIM_T, M2 = M1 / 2, M3 = M1 / 4;

  // workspace carve (floats)
  float* ws = (float*)d_ws;
  size_t off = 0;
  auto AL = [&](size_t n) { float* p = ws + off; off += n; return p; };
  float* x1    = AL((size_t)M1 * DIM_D);
  float* skip1 = AL((size_t)M1 * DIM_D);
  float* x2    = AL((size_t)M2 * DIM_D);
  float* skip2 = AL((size_t)M2 * DIM_D);
  float* x3    = AL((size_t)M3 * DIM_D);
  float* gbuf  = AL(DIM_B * 256);
  float* cbt   = AL(DIM_B * DIM_D);
  float* cb    = AL(DIM_B * DIM_D);
  float* cbc   = AL(DIM_B * DIM_D);
  float* t_y   = AL((size_t)M1 * DIM_D);
  float* t_q   = AL((size_t)M1 * DIM_D * 4);  // q|k|v|ctx; reused as ff / cat
  float* t_k   = t_q + (size_t)M1 * DIM_D;
  float* t_v   = t_k + (size_t)M1 * DIM_D;
  float* t_ctx = t_v + (size_t)M1 * DIM_D;
  float* t_attn = AL((size_t)M1 * DIM_D);
  float* t_qp  = AL((size_t)M1 * DIM_H * DIM_FD);
  float* t_kp  = AL((size_t)M1 * DIM_H * DIM_FD);
  float* t_z   = AL((size_t)M1 * DIM_H);
  float* t_ksum = AL(DIM_B * DIM_H * DIM_FD);
  float* t_kv  = AL((size_t)DIM_B * DIM_H * DIM_FD * DIM_DH);
  float* t_cat2 = AL((size_t)M2 * 2 * DIM_D);

  Temps t;
  t.y = t_y; t.q = t_q; t.k = t_k; t.v = t_v; t.ctx = t_ctx;
  t.attn = t_attn; t.ff = t_q; t.qp = t_qp; t.kp = t_kp; t.z = t_z;
  t.ksum = t_ksum; t.kv = t_kv; t.cbc = cbc; t.cb = cb;

  // --- global conditioning ---
  k_condmean<<<dim3((DIM_B * DIM_CD + 255) / 256), dim3(256), 0, stream>>>(
      f0, ph, sg, lg, gbuf);
  gemm(stream, gbuf, DIM_CD, cond1.w, DIM_D, cbt, DIM_D, DIM_B, DIM_D, DIM_CD,
       cond1.b, EP_RELU);
  gemm(stream, cbt, DIM_D, cond2.w, DIM_D, cb, DIM_D, DIM_B, DIM_D, DIM_D,
       cond2.b);

  // --- input projection + positional encoding ---
  gemm(stream, mel, DIM_NM, input_proj.w, DIM_D, x1, DIM_D, M1, DIM_D, DIM_NM,
       input_proj.b);
  {
    long long tot = (long long)M1 * DIM_D;
    k_posenc<<<dim3((unsigned)((tot + 255) / 256)), dim3(256), 0, stream>>>(
        x1, M1, DIM_T);
  }
  copycols(stream, x1, DIM_D, skip1, DIM_D, 0, M1, DIM_D);

  // --- level 1 ---
  for (int i = 0; i < 4; ++i) run_block(stream, x1, M1, DIM_T, blocks[i], t);

  // --- down1: maxpool + linear ---
  {
    long long tot = (long long)M2 * DIM_D;
    k_maxpool<<<dim3((unsigned)((tot + 255) / 256)), dim3(256), 0, stream>>>(
        x1, t_y, DIM_T / 2, DIM_D);
  }
  gemm(stream, t_y, DIM_D, down1.w, DIM_D, x2, DIM_D, M2, DIM_D, DIM_D,
       down1.b);

  // --- level 2 ---
  for (int i = 4; i < 8; ++i) run_block(stream, x2, M2, DIM_T / 2, blocks[i], t);
  copycols(stream, x2, DIM_D, skip2, DIM_D, 0, M2, DIM_D);

  // --- down2 ---
  {
    long long tot = (long long)M3 * DIM_D;
    k_maxpool<<<dim3((unsigned)((tot + 255) / 256)), dim3(256), 0, stream>>>(
        x2, t_y, DIM_T / 4, DIM_D);
  }
  gemm(stream, t_y, DIM_D, down2.w, DIM_D, x3, DIM_D, M3, DIM_D, DIM_D,
       down2.b);

  // --- level 3 ---
  for (int i = 8; i < 12; ++i) run_block(stream, x3, M3, DIM_T / 4, blocks[i], t);

  // --- up1: nearest x2 + linear, concat with skip2 ---
  {
    long long tot = (long long)M2 * DIM_D;
    k_upsample<<<dim3((unsigned)((tot + 255) / 256)), dim3(256), 0, stream>>>(
        x3, t_y, DIM_T / 2, DIM_D);
  }
  gemm(stream, t_y, DIM_D, up1.w, DIM_D, t_cat2, 2 * DIM_D, M2, DIM_D, DIM_D,
       up1.b);
  copycols(stream, skip2, DIM_D, t_cat2, 2 * DIM_D, DIM_D, M2, DIM_D);

  // --- up2: nearest x2 (width 2D) + linear 2D->D, concat with skip1 ---
  float* t_big  = t_q;                       // M1 x 1024
  float* t_cat1 = t_v;                       // M1 x 1024
  {
    long long tot = (long long)M1 * 2 * DIM_D;
    k_upsample<<<dim3((unsigned)((tot + 255) / 256)), dim3(256), 0, stream>>>(
        t_cat2, t_big, DIM_T, 2 * DIM_D);
  }
  gemm(stream, t_big, 2 * DIM_D, up2.w, DIM_D, t_cat1, 2 * DIM_D, M1, DIM_D,
       2 * DIM_D, up2.b);
  copycols(stream, skip1, DIM_D, t_cat1, 2 * DIM_D, DIM_D, M1, DIM_D);

  // --- output projection + mel residual ---
  gemm(stream, t_cat1, 2 * DIM_D, outp.w, DIM_NM, (float*)d_out, DIM_NM, M1,
       DIM_NM, 2 * DIM_D, outp.b, 0, 1.f, mel, DIM_NM);
}
